// PerformerAttention_6305011991167
// MI455X (gfx1250) — matrix operand order repaired and checked by reading, not compiled, not run on hardware
//
#include <hip/hip_runtime.h>
#include <math.h>

#define DEV_INLINE __device__ __forceinline__

typedef __attribute__((ext_vector_type(16))) _Float16 v16h;
typedef __attribute__((ext_vector_type(8)))  float    v8f;

// ---------------------------------------------------------------------------
// WMMA helpers (gfx1250, wave32): D(16x16 f32) = A(16x32 f16) x B(32x16 f16) + C
// A and B fragments use contiguous k order: element s of lane half h holds k = 16h + s.
// ---------------------------------------------------------------------------
DEV_INLINE v8f wmma16(v16h a, v16h b, v8f c) {
  return __builtin_amdgcn_wmma_f32_16x16x32_f16(false, a, false, b, (short)0, c,
                                                false, false);
}

DEV_INLINE v16h load_a_frag(const _Float16* lds, int stride, int row0, int k0,
                            int lane) {
  int row = row0 + (lane & 15);
  int kg  = (lane >> 4) & 1;
  const _Float16* p = lds + row * stride + k0;
  v16h f;
#pragma unroll
  for (int i = 0; i < 8; ++i) f[i] = p[16 * kg + i];
#pragma unroll
  for (int i = 0; i < 8; ++i) f[8 + i] = p[16 * kg + 8 + i];
  return f;
}

DEV_INLINE v16h load_b_frag(const _Float16* ldsBT, int stride, int col0, int k0,
                            int lane) {
  int col = col0 + (lane & 15);
  int kb  = k0 + 16 * ((lane >> 4) & 1);
  const _Float16* p = ldsBT + col * stride + kb;
  v16h f;
#pragma unroll
  for (int i = 0; i < 16; ++i) f[i] = p[i];
  return f;
}

// ---------------------------------------------------------------------------
// gfx1250 async global->LDS copy (ASYNCcnt-tracked DMA, bypasses VGPRs).
// LDS operand is the low-32 bits of the generic shared pointer (flat aperture
// rule). No builtin exposed -> inline asm.
// ---------------------------------------------------------------------------
DEV_INLINE void async_ld_b128(void* lds_dst, const void* gsrc) {
  unsigned int lds_addr = (unsigned int)(unsigned long long)lds_dst;
  asm volatile("global_load_async_to_lds_b128 %0, %1, off"
               :
               : "v"(lds_addr), "v"(gsrc)
               : "memory");
}
DEV_INLINE void wait_async0() {
  asm volatile("s_wait_asynccnt 0x0" ::: "memory");
}

// ---------------------------------------------------------------------------
// f32 -> f16 conversion
// ---------------------------------------------------------------------------
__global__ void performer_f32_to_f16(const float* __restrict__ in,
                                     _Float16* __restrict__ out, int n) {
  int i = (blockIdx.x * blockDim.x + threadIdx.x) * 4;
  if (i + 3 < n) {
    float4 x = *(const float4*)(in + i);
    out[i + 0] = (_Float16)x.x;
    out[i + 1] = (_Float16)x.y;
    out[i + 2] = (_Float16)x.z;
    out[i + 3] = (_Float16)x.w;
  }
}

// ---------------------------------------------------------------------------
// NT GEMM: C[Mr x Nc] = A[Mr x K] * W[Nc x K]^T + bias
// Block tile 128x64, 8 waves (4x2), 32x32 per wave, K step 32.
// Double-buffered LDS, async global->LDS staging overlapped with WMMA.
// ---------------------------------------------------------------------------
__global__ void __launch_bounds__(256) performer_gemm_nt(
    const _Float16* __restrict__ A, const _Float16* __restrict__ W,
    const float* __restrict__ bias, _Float16* __restrict__ out16,
    float* __restrict__ out32, int Mr, int Nc, int K) {
  __shared__ __align__(16) _Float16 sA[2][128 * 40];
  __shared__ __align__(16) _Float16 sB[2][64 * 40];
  const int tid = threadIdx.x, lane = tid & 31, wave = tid >> 5;
  const int wm = wave & 3, wn = wave >> 2;
  const int mbase = blockIdx.x * 128;
  const int nbase = blockIdx.y * 64;

  // Per-thread staging slots (16B chunks)
  const int ar0 = (tid * 2) >> 2, ak0 = ((tid * 2) & 3) * 8;
  const int ar1 = (tid * 2 + 1) >> 2, ak1 = ((tid * 2 + 1) & 3) * 8;
  const int br = tid >> 2, bk = (tid & 3) * 8;

  auto stage = [&](int buf, int k0) {
    async_ld_b128(&sA[buf][ar0 * 40 + ak0],
                  A + (size_t)(mbase + ar0) * K + k0 + ak0);
    async_ld_b128(&sA[buf][ar1 * 40 + ak1],
                  A + (size_t)(mbase + ar1) * K + k0 + ak1);
    async_ld_b128(&sB[buf][br * 40 + bk],
                  W + (size_t)(nbase + br) * K + k0 + bk);
  };

  v8f acc[2][2] = {};
  stage(0, 0);
  int cur = 0;
  for (int k0 = 0; k0 < K; k0 += 32) {
    wait_async0();          // my async copies into sX[cur] have landed
    __syncthreads();        // everyone's copies landed; prev reads finished
    if (k0 + 32 < K) {
      stage(cur ^ 1, k0 + 32);
      if (k0 + 64 < K)      // long-range hint (global_prefetch_b8)
        __builtin_prefetch(A + (size_t)(mbase + ar0) * K + k0 + 64, 0, 1);
    }
    const _Float16* a = sA[cur];
    const _Float16* bt = sB[cur];
#pragma unroll
    for (int i = 0; i < 2; ++i) {
      v16h af = load_a_frag(a, 40, wm * 32 + i * 16, 0, lane);
#pragma unroll
      for (int j = 0; j < 2; ++j) {
        v16h bf = load_b_frag(bt, 40, wn * 32 + j * 16, 0, lane);
        acc[i][j] = wmma16(af, bf, acc[i][j]);
      }
    }
    cur ^= 1;
  }
#pragma unroll
  for (int i = 0; i < 2; ++i)
#pragma unroll
    for (int j = 0; j < 2; ++j) {
      int col = nbase + wn * 32 + j * 16 + (lane & 15);
      float bv = bias ? bias[col] : 0.0f;
#pragma unroll
      for (int r = 0; r < 8; ++r) {
        int row = mbase + wm * 32 + i * 16 + r + 8 * (lane >> 4);
        float vv = acc[i][j][r] + bv;
        if (out16) out16[(size_t)row * Nc + col] = (_Float16)vv;
        if (out32) out32[(size_t)row * Nc + col] = vv;
      }
    }
}

// ---------------------------------------------------------------------------
// Feature kernel: per (b,h), data_dash = s * (x @ proj^T), diag = 0.5*s^2*|x|^2
//   query: q' = exp(dash - rowmax(dash) - diag)/sqrt(M) -> [bh][n][256]
//   key  : k' = exp(dash - diag)/sqrt(M) stored TRANSPOSED -> [bh][256][n]
// Workgroup: 128 rows x all 256 feature cols (row-max stays in-workgroup).
// ---------------------------------------------------------------------------
__global__ void __launch_bounds__(256) performer_features(
    const _Float16* __restrict__ act,   // [B*N, D] f16
    const _Float16* __restrict__ proj,  // [256, 64] f16
    _Float16* __restrict__ outQ,        // [bh][4096][256]
    _Float16* __restrict__ outKT,       // [bh][256][4096]
    int is_query) {
  const int Nn = 4096, Dd = 1024, HD = 64, Mf = 256;
  __shared__ __align__(16) _Float16 sP[256 * 72];
  __shared__ __align__(16) _Float16 sA[128 * 72];
  __shared__ float sDiag[128];
  const int tid = threadIdx.x, lane = tid & 31, wave = tid >> 5;
  const int bh = blockIdx.y, b = bh >> 4, h = bh & 15;
  const int nbase = blockIdx.x * 128;

  // Stage proj (256x64) and x tile (128x64) via async DMA: 16B chunks
#pragma unroll
  for (int c = 0; c < 8; ++c) {
    int chunk = tid * 8 + c;
    int row = chunk >> 3;
    int kk = (chunk & 7) * 8;
    async_ld_b128(&sP[row * 72 + kk], proj + row * 64 + kk);
  }
#pragma unroll
  for (int c = 0; c < 4; ++c) {
    int chunk = tid * 4 + c;
    int row = chunk >> 3;
    int kk = (chunk & 7) * 8;
    async_ld_b128(&sA[row * 72 + kk],
                  act + (size_t)(b * Nn + nbase + row) * Dd + h * HD + kk);
  }
  wait_async0();
  __syncthreads();
  if (tid < 128) {
    float ss = 0.0f;
    for (int d = 0; d < 64; ++d) {
      float x = (float)sA[tid * 72 + d];
      ss += x * x;
    }
    sDiag[tid] = 0.5f * 0.125f * ss;  // s^2 = 64^-0.5 = 1/8
  }
  __syncthreads();

  v8f acc[16] = {};
#pragma unroll
  for (int k0 = 0; k0 < 64; k0 += 32) {
    v16h af = load_a_frag(sA, 72, wave * 16, k0, lane);
#pragma unroll
    for (int t = 0; t < 16; ++t) {
      v16h bf = load_b_frag(sP, 72, t * 16, k0, lane);
      acc[t] = wmma16(af, bf, acc[t]);
    }
  }

  const float s = 0.3535533905932738f;  // 64^{-1/4}
  const float invSqrtM = 0.0625f;       // 1/sqrt(256)
  if (is_query) {
#pragma unroll
    for (int r = 0; r < 8; ++r) {
      float mx = -1e30f;
#pragma unroll
      for (int t = 0; t < 16; ++t) mx = fmaxf(mx, acc[t][r]);
      mx = fmaxf(mx, __shfl_xor(mx, 1));
      mx = fmaxf(mx, __shfl_xor(mx, 2));
      mx = fmaxf(mx, __shfl_xor(mx, 4));
      mx = fmaxf(mx, __shfl_xor(mx, 8));
      int rloc = wave * 16 + r + 8 * (lane >> 4);
      float dg = sDiag[rloc];
      int n = nbase + rloc;
#pragma unroll
      for (int t = 0; t < 16; ++t) {
        int m = t * 16 + (lane & 15);
        float vv = expf(s * (acc[t][r] - mx) - dg) * invSqrtM;
        outQ[((size_t)bh * Nn + n) * Mf + m] = (_Float16)vv;
      }
    }
  } else {
#pragma unroll
    for (int r = 0; r < 8; ++r) {
      int rloc = wave * 16 + r + 8 * (lane >> 4);
      float dg = sDiag[rloc];
      int n = nbase + rloc;
#pragma unroll
      for (int t = 0; t < 16; ++t) {
        int m = t * 16 + (lane & 15);
        float vv = expf(s * acc[t][r] - dg) * invSqrtM;
        outKT[((size_t)bh * Mf + m) * Nn + n] = (_Float16)vv;
      }
    }
  }
}

// ---------------------------------------------------------------------------
// kv kernel: per (b,h): kv_ext[256 x 80] = kT[256 x 4096] @ [v | 1 | 0pad]
// Column 64 accumulates k_sum (ones column); cols 65..79 are zero padding.
// ---------------------------------------------------------------------------
__global__ void __launch_bounds__(256) performer_kv(
    const _Float16* __restrict__ kT,    // [bh][256][4096]
    const _Float16* __restrict__ vact,  // [B*N, D]
    _Float16* __restrict__ kvext) {     // [bh][256][80]
  const int Nn = 4096, Dd = 1024, HD = 64, Mf = 256, CE = 80;
  __shared__ __align__(16) _Float16 sA[256 * 40];
  __shared__ __align__(16) _Float16 sB[80 * 40];
  const int tid = threadIdx.x, lane = tid & 31, wave = tid >> 5;
  const int bh = blockIdx.x, b = bh >> 4, h = bh & 15;
  const _Float16* kTb = kT + (size_t)bh * Mf * Nn;

  v8f acc[2][5] = {};
  for (int n0 = 0; n0 < Nn; n0 += 32) {
    // A tile (256x32) via async DMA: 4 x 16B chunks per thread
#pragma unroll
    for (int c = 0; c < 4; ++c) {
      int chunk = tid * 4 + c;
      int row = chunk >> 2;
      int kk = (chunk & 3) * 8;
      async_ld_b128(&sA[row * 40 + kk], kTb + (size_t)row * Nn + n0 + kk);
    }
    // B^T in LDS: sB[c][k] = v[n0+k][c] (c<64), ones (c==64), zero pad
#pragma unroll
    for (int e = 0; e < 10; ++e) {
      int idx = tid * 10 + e;
      int cc = idx >> 5;
      int kk = idx & 31;
      _Float16 vv;
      if (cc < 64)
        vv = vact[(size_t)(b * Nn + n0 + kk) * Dd + h * HD + cc];
      else if (cc == 64)
        vv = (_Float16)1.0f;
      else
        vv = (_Float16)0.0f;
      sB[cc * 40 + kk] = vv;
    }
    wait_async0();
    __syncthreads();
#pragma unroll
    for (int i = 0; i < 2; ++i) {
      v16h af = load_a_frag(sA, 40, wave * 32 + i * 16, 0, lane);
#pragma unroll
      for (int j = 0; j < 5; ++j) {
        v16h bf = load_b_frag(sB, 40, j * 16, 0, lane);
        acc[i][j] = wmma16(af, bf, acc[i][j]);
      }
    }
    __syncthreads();
  }
  _Float16* outb = kvext + (size_t)bh * Mf * CE;
#pragma unroll
  for (int i = 0; i < 2; ++i)
#pragma unroll
    for (int j = 0; j < 5; ++j) {
      int col = j * 16 + (lane & 15);
#pragma unroll
      for (int r = 0; r < 8; ++r) {
        int row = wave * 32 + i * 16 + r + 8 * (lane >> 4);
        outb[row * CE + col] = (_Float16)acc[i][j][r];
      }
    }
}

// ---------------------------------------------------------------------------
// out kernel: per (b,h): out_ext[4096 x 80] = q'[4096 x 256] @ kv_ext[256 x 80]
// Column 64 is the normalizer; broadcast per-row, divide, write merged heads.
// ---------------------------------------------------------------------------
__global__ void __launch_bounds__(256) performer_out(
    const _Float16* __restrict__ qp,     // [bh][4096][256]
    const _Float16* __restrict__ kvext,  // [bh][256][80]
    _Float16* __restrict__ yattn) {      // [B*N, D]
  const int Nn = 4096, Dd = 1024, HD = 64, Mf = 256, CE = 80;
  __shared__ __align__(16) _Float16 sA[128 * 40];
  __shared__ __align__(16) _Float16 sB[80 * 40];
  const int tid = threadIdx.x, lane = tid & 31, wave = tid >> 5;
  const int bh = blockIdx.y, b = bh >> 4, h = bh & 15;
  const int nbase = blockIdx.x * 128;
  const _Float16* qpb = qp + (size_t)bh * Nn * Mf;
  const _Float16* kvb = kvext + (size_t)bh * Mf * CE;

  v8f acc[5] = {};
  for (int k0 = 0; k0 < Mf; k0 += 32) {
#pragma unroll
    for (int c = 0; c < 2; ++c) {
      int chunk = tid * 2 + c;
      int row = chunk >> 2;
      int kk = (chunk & 3) * 8;
      async_ld_b128(&sA[row * 40 + kk],
                    qpb + (size_t)(nbase + row) * Mf + k0 + kk);
    }
#pragma unroll
    for (int e = 0; e < 10; ++e) {
      int idx = tid * 10 + e;
      int cc = idx >> 5;
      int kk = idx & 31;
      sB[cc * 40 + kk] = kvb[(k0 + kk) * CE + cc];
    }
    wait_async0();
    __syncthreads();
    v16h af = load_a_frag(sA, 40, wave * 16, 0, lane);
#pragma unroll
    for (int j = 0; j < 5; ++j) {
      v16h bf = load_b_frag(sB, 40, j * 16, 0, lane);
      acc[j] = wmma16(af, bf, acc[j]);
    }
    __syncthreads();
  }
#pragma unroll
  for (int r = 0; r < 8; ++r) {
    float norm = __shfl(acc[4][r], lane & 16);
    norm = fmaxf(norm, 1e-6f);
    int n = nbase + wave * 16 + r + 8 * (lane >> 4);
#pragma unroll
    for (int j = 0; j < 4; ++j) {
      int d = j * 16 + (lane & 15);
      yattn[(size_t)(b * Nn + n) * Dd + h * HD + d] =
          (_Float16)(acc[j][r] / norm);
    }
  }
}

// ---------------------------------------------------------------------------
// Host driver
// ---------------------------------------------------------------------------
extern "C" void kernel_launch(void* const* d_in, const int* in_sizes, int n_in,
                              void* d_out, int out_size, void* d_ws,
                              size_t ws_size, hipStream_t stream) {
  (void)in_sizes; (void)n_in; (void)out_size; (void)ws_size;
  const float* q    = (const float*)d_in[0];
  const float* k    = (const float*)d_in[1];
  const float* v    = (const float*)d_in[2];
  const float* Wq   = (const float*)d_in[3];
  const float* bq   = (const float*)d_in[4];
  const float* Wk   = (const float*)d_in[5];
  const float* bk   = (const float*)d_in[6];
  const float* Wv   = (const float*)d_in[7];
  const float* bv   = (const float*)d_in[8];
  const float* Wo   = (const float*)d_in[9];
  const float* bo   = (const float*)d_in[10];
  const float* proj = (const float*)d_in[11];

  const int Bb = 4, Nn = 4096, Dd = 1024, Mf = 256, HD = 64;
  const size_t NX  = (size_t)Bb * Nn * Dd;       // 16,777,216
  const size_t NW  = (size_t)Dd * Dd;            // 1,048,576
  const size_t NP  = (size_t)Mf * HD;            // 16,384
  const size_t NQP = (size_t)Bb * 16 * Nn * Mf;  // 67,108,864
  const size_t NKV = (size_t)Bb * 16 * Mf * 80;

  char* base = (char*)d_ws;
  size_t off = 0;
  auto a16 = [&](size_t elems) -> _Float16* {
    _Float16* p = (_Float16*)(base + off);
    off += ((elems * 2) + 255) & ~(size_t)255;
    return p;
  };
  _Float16* q16  = a16(NX);
  _Float16* k16  = a16(NX);
  _Float16* v16  = a16(NX);
  _Float16* Wq16 = a16(NW);
  _Float16* Wk16 = a16(NW);
  _Float16* Wv16 = a16(NW);
  _Float16* Wo16 = a16(NW);
  _Float16* pj16 = a16(NP);
  _Float16* Qact = a16(NX);
  _Float16* Kact = a16(NX);
  _Float16* Vact = a16(NX);
  _Float16* qpr  = a16(NQP);
  _Float16* kTp  = a16(NQP);
  _Float16* kvex = a16(NKV);
  _Float16* yat  = a16(NX);

  dim3 blk(256);
  auto cvt = [&](const float* in, _Float16* outp, size_t n) {
    int nth = (int)(n / 4);
    performer_f32_to_f16<<<dim3((nth + 255) / 256), blk, 0, stream>>>(
        in, outp, (int)n);
  };
  cvt(q, q16, NX);
  cvt(k, k16, NX);
  cvt(v, v16, NX);
  cvt(Wq, Wq16, NW);
  cvt(Wk, Wk16, NW);
  cvt(Wv, Wv16, NW);
  cvt(Wo, Wo16, NW);
  cvt(proj, pj16, NP);

  dim3 gproj(16384 / 128, 1024 / 64);
  performer_gemm_nt<<<gproj, blk, 0, stream>>>(q16, Wq16, bq, Qact,
                                               (float*)nullptr, 16384, 1024, 1024);
  performer_gemm_nt<<<gproj, blk, 0, stream>>>(k16, Wk16, bk, Kact,
                                               (float*)nullptr, 16384, 1024, 1024);
  performer_gemm_nt<<<gproj, blk, 0, stream>>>(v16, Wv16, bv, Vact,
                                               (float*)nullptr, 16384, 1024, 1024);

  dim3 gfeat(4096 / 128, 64);
  performer_features<<<gfeat, blk, 0, stream>>>(Qact, pj16, qpr, kTp, 1);
  performer_features<<<gfeat, blk, 0, stream>>>(Kact, pj16, qpr, kTp, 0);

  performer_kv<<<dim3(64), blk, 0, stream>>>(kTp, Vact, kvex);
  performer_out<<<dim3(4096 / 128, 64), blk, 0, stream>>>(qpr, kvex, yat);

  performer_gemm_nt<<<gproj, blk, 0, stream>>>(yat, Wo16, bo,
                                               (_Float16*)nullptr, (float*)d_out,
                                               16384, 1024, 1024);
}
